// AttentivePoolingLayer_1726576855958
// MI455X (gfx1250) — matrix operand order repaired and checked by reading, not compiled, not run on hardware
//
#include <hip/hip_runtime.h>

// ---------------------------------------------------------------------------
// Types for CDNA5 WMMA / TDM
// ---------------------------------------------------------------------------
typedef __attribute__((ext_vector_type(16))) __bf16        v16bf;
typedef __attribute__((ext_vector_type(8)))  float         v8f;
typedef __attribute__((ext_vector_type(8)))  unsigned int  v8u;
typedef __attribute__((ext_vector_type(4)))  unsigned int  v4u;
typedef __attribute__((ext_vector_type(8)))  int           v8i;
typedef __attribute__((ext_vector_type(4)))  int           v4i;

#define NA   16      // batch a
#define NB   16      // batch b
#define DIM  512     // dA == dB
#define SEQ  512     // sA == sB

// ---------------------------------------------------------------------------
// helpers
// ---------------------------------------------------------------------------
__device__ __forceinline__ unsigned short f2bf(float f) {
    unsigned int u = __float_as_uint(f);
    u += 0x7FFFu + ((u >> 16) & 1u);          // round-to-nearest-even
    return (unsigned short)(u >> 16);
}

__device__ __forceinline__ void atomicMaxF(float* addr, float val) {
    unsigned int* ua = (unsigned int*)addr;
    unsigned int old = *ua, assumed;
    do {
        assumed = old;
        if (__uint_as_float(assumed) >= val) break;
        old = atomicCAS(ua, assumed, __float_as_uint(val));
    } while (old != assumed);
}

// ---------------------------------------------------------------------------
// TDM 2D descriptor build (D# group0/group1 per ISA 8.3/8.4) + load/store.
// dsizeLog: 0=1B 1=2B 2=4B. tileW/strideElems in elements of that size.
// ---------------------------------------------------------------------------
__device__ __forceinline__ void tdm_desc(v4u& g0, v8i& g1,
                                         unsigned ldsOff, const void* gptr,
                                         unsigned tileW, unsigned tileH,
                                         unsigned strideElems, unsigned dsizeLog)
{
    unsigned long long ga = (unsigned long long)gptr;
    g0[0] = 1u;                                    // count=1 (valid descriptor)
    g0[1] = ldsOff;                                // lds_addr (bytes)
    g0[2] = (unsigned)ga;                          // global_addr[31:0]
    g0[3] = (unsigned)(ga >> 32) | (2u << 30);     // global_addr[56:32] | type=2

    const unsigned td0 = strideElems;              // tensor_dim0 (never OOB here)
    const unsigned td1 = 0x7FFFFFFFu;              // tensor_dim1: huge
    g1[0] = (int)(dsizeLog << 16);                 // wg_mask=0 | data_size
    g1[1] = (int)((td0 & 0xFFFFu) << 16);          // tensor_dim0[15:0] (<<16)
    g1[2] = (int)((td0 >> 16) | ((td1 & 0xFFFFu) << 16));
    g1[3] = (int)((td1 >> 16) | (tileW << 16));    // tensor_dim1 hi | tile_dim0
    g1[4] = (int)tileH;                            // tile_dim1 | tile_dim2=0
    g1[5] = (int)strideElems;                      // tensor_dim0_stride[31:0]
    g1[6] = 0;
    g1[7] = 0;
}

__device__ __forceinline__ void tdm_load_2d(unsigned ldsOff, const void* gptr,
                                            unsigned tileW, unsigned tileH,
                                            unsigned strideElems, unsigned dsizeLog)
{
    v4u g0; v8i g1;
    tdm_desc(g0, g1, ldsOff, gptr, tileW, tileH, strideElems, dsizeLog);
    const v4i z4 = {0, 0, 0, 0};
    const v8i z8 = {0, 0, 0, 0, 0, 0, 0, 0};
    __builtin_amdgcn_tensor_load_to_lds(g0, g1, z4, z4, z8, 0);
}

__device__ __forceinline__ void tdm_store_2d(unsigned ldsOff, void* gptr,
                                             unsigned tileW, unsigned tileH,
                                             unsigned strideElems, unsigned dsizeLog)
{
    v4u g0; v8i g1;
    tdm_desc(g0, g1, ldsOff, gptr, tileW, tileH, strideElems, dsizeLog);
    const v4i z4 = {0, 0, 0, 0};
    const v8i z8 = {0, 0, 0, 0, 0, 0, 0, 0};
    __builtin_amdgcn_tensor_store_from_lds(g0, g1, z4, z4, z8, 0);
}

// ---------------------------------------------------------------------------
// 1a. transpose-convert A: (a, d, s) f32 -> At (a, s, d) bf16 row-major
// ---------------------------------------------------------------------------
__global__ void k_transposeA(const float* __restrict__ A, unsigned short* __restrict__ At) {
    __shared__ float t[32 * 33];
    const int a = blockIdx.z, tileD = blockIdx.y * 32, tileS = blockIdx.x * 32;
    const int tx = threadIdx.x, ty = threadIdx.y;
#pragma unroll
    for (int i = 0; i < 4; ++i) {
        int dl = ty + i * 8;
        t[dl * 33 + tx] = A[((size_t)a * DIM + tileD + dl) * SEQ + tileS + tx];
    }
    __syncthreads();
#pragma unroll
    for (int i = 0; i < 4; ++i) {
        int sl = ty + i * 8;
        At[((size_t)a * SEQ + tileS + sl) * DIM + tileD + tx] = f2bf(t[tx * 33 + sl]);
    }
}

// ---------------------------------------------------------------------------
// 1b. swizzle-convert: dst word (bb, n, kp) = bf16(src[bb][2kp][n])
//                                           | bf16(src[bb][2kp+1][n]) << 16
// Layout is (n-major, kp fastest): a TDM tile load of 16 words x 128 rows
// (stride 256 words) lands in LDS as [n][kp] -- the B-fragment
// order used here (k = 16h + s), so each fragment is 8 *contiguous* words (two ds_load_b128).
// ---------------------------------------------------------------------------
__global__ void k_swzconvert(const float* __restrict__ src, unsigned int* __restrict__ dst,
                             int nWords) {
    int w = blockIdx.x * 256 + threadIdx.x;
    if (w >= nWords) return;
    int kp = w & 255;
    int n  = (w >> 8) & (DIM - 1);
    int bb = w >> 17;
    size_t base = ((size_t)bb * DIM + 2 * kp) * DIM + n;
    unsigned lo = f2bf(src[base]);
    unsigned hi = f2bf(src[base + DIM]);
    dst[w] = lo | (hi << 16);
}

// ---------------------------------------------------------------------------
// 2. fill float buffer with -inf
// ---------------------------------------------------------------------------
__global__ void k_fillneg(float* __restrict__ dst, int n) {
    int i = blockIdx.x * 256 + threadIdx.x;
    if (i < n) dst[i] = -__builtin_inff();
}

// ---------------------------------------------------------------------------
// Core 128x128 (K=512) bf16 WMMA tile GEMM with TDM double buffering.
//   Ag: bf16 element (m,k) at Ag[m*512 + k]       (row-major)
//   Bg: u32 word (n,kp)    at Bg[n*256 + kp]      (n-major pre-swizzle)
// smem: 32KB = sA[2] (16KB) + sB[2] (16KB).
// 256 threads = 8 waves, each wave owns a 32x64 sub-tile = 2x4 C tiles.
// Wave 0 drives the TDM engine; k-step i+1 streams in while i computes.
// ---------------------------------------------------------------------------
__device__ __forceinline__ void mm_tile_128x128(
        const unsigned short* __restrict__ Ag,
        const unsigned int*  __restrict__ Bg,
        unsigned char* smem,
        v8f acc[2][4], int tid)
{
    const int lane = tid & 31;
    const int l16  = lane & 15;
    const int h    = lane >> 4;                  // lane half
    const int wave = tid >> 5;
    const int m0w  = (wave >> 1) * 32;           // wave row base
    const int n0w  = (wave & 1) * 64;            // wave col base
    const int NIT  = DIM / 32;

    unsigned short* sA0 = (unsigned short*)smem;              // 128x32 bf16
    unsigned short* sA1 = sA0 + 128 * 32;
    unsigned int*   sB0 = (unsigned int*)(smem + 16384);      // 128x16 words
    unsigned int*   sB1 = sB0 + 128 * 16;

    const unsigned ldsA0 = (unsigned)(size_t)sA0;
    const unsigned ldsA1 = (unsigned)(size_t)sA1;
    const unsigned ldsB0 = (unsigned)(size_t)sB0;
    const unsigned ldsB1 = (unsigned)(size_t)sB1;

    if (tid < 32) {   // wave 0: prime buffer 0
        tdm_load_2d(ldsA0, Ag, 32, 128, DIM, 1);       // 128 rows x 32 bf16
        tdm_load_2d(ldsB0, Bg, 16, 128, DIM / 2, 2);   // 128 rows x 16 words
    }

    for (int i = 0; i < NIT; ++i) {
        const int cur = i & 1;
        if (tid < 32) {
            if (i + 1 < NIT) {   // prefetch next k-step into other buffer
                const int kn = (i + 1) * 32;
                tdm_load_2d(cur ? ldsA0 : ldsA1, Ag + kn,         32, 128, DIM, 1);
                tdm_load_2d(cur ? ldsB0 : ldsB1, Bg + (kn >> 1),  16, 128, DIM / 2, 2);
                __builtin_amdgcn_s_wait_tensorcnt(2);   // current pair landed
            } else {
                __builtin_amdgcn_s_wait_tensorcnt(0);
            }
        }
        __syncthreads();   // publish buf[cur] to all waves

        const unsigned short* sA = cur ? sA1 : sA0;
        const unsigned int*   sB = cur ? sB1 : sB0;

        // --- A fragments (16x32 bf16, contiguous k = 16h + s; 2x b128 each) ---
        v16bf af[2];
#pragma unroll
        for (int r = 0; r < 2; ++r) {
            const uint4* p = (const uint4*)(sA + (m0w + r * 16 + l16) * 32 + 16 * h);
            uint4 q0 = p[0], q1 = p[1];
            v8u t;
            t[0] = q0.x; t[1] = q0.y; t[2] = q0.z; t[3] = q0.w;
            t[4] = q1.x; t[5] = q1.y; t[6] = q1.z; t[7] = q1.w;
            af[r] = __builtin_bit_cast(v16bf, t);
        }
        // --- B fragments (32x16 bf16; contiguous [n][kp]: 2x b128 each) ---
        v16bf bfr[4];
#pragma unroll
        for (int c = 0; c < 4; ++c) {
            const uint4* p = (const uint4*)(sB + (n0w + c * 16 + l16) * 16 + 8 * h);
            uint4 q0 = p[0], q1 = p[1];
            v8u t;
            t[0] = q0.x; t[1] = q0.y; t[2] = q0.z; t[3] = q0.w;
            t[4] = q1.x; t[5] = q1.y; t[6] = q1.z; t[7] = q1.w;
            bfr[c] = __builtin_bit_cast(v16bf, t);
        }
        // --- 8 WMMAs ---
#pragma unroll
        for (int r = 0; r < 2; ++r)
#pragma unroll
            for (int c = 0; c < 4; ++c)
                acc[r][c] = __builtin_amdgcn_wmma_f32_16x16x32_bf16(
                        false, af[r], false, bfr[c], (short)0, acc[r][c], false, false);

        __syncthreads();   // all reads of buf[cur] done before TDM overwrites it
    }
}

// ---------------------------------------------------------------------------
// 3. proj[a] = At[a] (sA x dA) @ U (dA x dB), bf16 out via TDM store.
//    C tile staged in LDS (reusing GEMM staging space) -> one
//    tensor_store_from_lds per block.
// ---------------------------------------------------------------------------
__global__ void k_gemm_proj(const unsigned short* __restrict__ At,
                            const unsigned int* __restrict__ Usw,
                            unsigned short* __restrict__ Pr)
{
    __shared__ __align__(16) unsigned char smem[32768];
    const int tid = threadIdx.x;
    const int a = blockIdx.y;
    const int tileS = (blockIdx.x >> 2) * 128;
    const int tileN = (blockIdx.x & 3) * 128;

    v8f acc[2][4] = {};
    mm_tile_128x128(At + (size_t)a * SEQ * DIM + (size_t)tileS * DIM,
                    Usw + tileN * (DIM / 2), smem, acc, tid);

    const int lane = tid & 31, wave = tid >> 5;
    const int l16 = lane & 15, h = lane >> 4;
    const int m0w = (wave >> 1) * 32, n0w = (wave & 1) * 64;

    // stage C tile (128x128 bf16, row-major) in the now-idle LDS
    unsigned short* cT = (unsigned short*)smem;
#pragma unroll
    for (int r = 0; r < 2; ++r)
#pragma unroll
        for (int c = 0; c < 4; ++c) {
            const int nl = n0w + c * 16 + l16;
#pragma unroll
            for (int vr = 0; vr < 8; ++vr) {
                const int ml = m0w + r * 16 + vr + 8 * h;
                cT[ml * 128 + nl] = f2bf(acc[r][c][vr]);
            }
        }
    __syncthreads();

    if (tid < 32) {
        unsigned short* dst = Pr + ((size_t)a * SEQ + tileS) * DIM + tileN;
        tdm_store_2d((unsigned)(size_t)cT, dst, 128, 128, DIM, 1);
        // S_ENDPGM performs an implicit wait-idle, draining TENSORcnt.
    }
}

// ---------------------------------------------------------------------------
// 4. align tile: S = tanh(proj[a] @ B[b]) + mask[a]; fused row/col max
//    with shfl-xor tree reductions (rows: within 16-lane group; cols:
//    across lane halves), then a handful of single-lane LDS atomics.
// ---------------------------------------------------------------------------
__global__ void k_align(const unsigned short* __restrict__ Pr,
                        const unsigned int* __restrict__ Bsw,
                        const float* __restrict__ msk,
                        float* __restrict__ mA, float* __restrict__ mB)
{
    __shared__ __align__(16) unsigned char smem[32768];
    __shared__ float sRowMax[128];
    __shared__ float sColMax[128];

    const int tid = threadIdx.x;
    const int a = blockIdx.z, b = blockIdx.y;
    const int tileS = (blockIdx.x >> 2) * 128;
    const int tileT = (blockIdx.x & 3) * 128;

    if (tid < 128) sRowMax[tid] = -__builtin_inff();
    else           sColMax[tid - 128] = -__builtin_inff();

    v8f acc[2][4] = {};
    mm_tile_128x128(Pr + (size_t)a * SEQ * DIM + (size_t)tileS * DIM,
                    Bsw + ((size_t)b * SEQ + tileT) * (DIM / 2),
                    smem, acc, tid);

    const int lane = tid & 31, wave = tid >> 5;
    const int l16 = lane & 15, h = lane >> 4;
    const int m0w = (wave >> 1) * 32, n0w = (wave & 1) * 64;

    float rmax[2][8];
    float cmax[4];
#pragma unroll
    for (int r = 0; r < 2; ++r)
#pragma unroll
        for (int vr = 0; vr < 8; ++vr) rmax[r][vr] = -__builtin_inff();
#pragma unroll
    for (int c = 0; c < 4; ++c) cmax[c] = -__builtin_inff();

#pragma unroll
    for (int r = 0; r < 2; ++r)
#pragma unroll
        for (int c = 0; c < 4; ++c) {
            const int nl = n0w + c * 16 + l16;
#pragma unroll
            for (int vr = 0; vr < 8; ++vr) {
                const int ml = m0w + r * 16 + vr + 8 * h;
                float v = tanhf(acc[r][c][vr]) +
                          msk[((size_t)a * SEQ + tileS + ml) * SEQ + tileT + nl];
                rmax[r][vr] = fmaxf(rmax[r][vr], v);
                cmax[c]     = fmaxf(cmax[c], v);
            }
        }

    // rows: reduce across the 16 lanes of this half (cols), 1 atomic per row
#pragma unroll
    for (int r = 0; r < 2; ++r)
#pragma unroll
        for (int vr = 0; vr < 8; ++vr) {
            float v = rmax[r][vr];
#pragma unroll
            for (int off = 8; off > 0; off >>= 1)
                v = fmaxf(v, __shfl_xor(v, off, 32));
            if (l16 == 0)
                atomicMaxF(&sRowMax[m0w + r * 16 + 8 * h + vr], v);
        }
    // cols: combine the two lane halves, 1 atomic per col
#pragma unroll
    for (int c = 0; c < 4; ++c) {
        float v = fmaxf(cmax[c], __shfl_xor(cmax[c], 16, 32));
        if (h == 0)
            atomicMaxF(&sColMax[n0w + c * 16 + l16], v);
    }
    __syncthreads();

    const int pair = a * NB + b;
    if (tid < 128) atomicMaxF(&mA[(size_t)pair * SEQ + tileS + tid], sRowMax[tid]);
    else           atomicMaxF(&mB[(size_t)pair * SEQ + tileT + tid - 128], sColMax[tid - 128]);
}

// ---------------------------------------------------------------------------
// 5. softmax over 512, grid (256 rows, 2 arrays), block 256
// ---------------------------------------------------------------------------
__global__ void k_softmax(const float* __restrict__ mA, const float* __restrict__ mB,
                          float* __restrict__ scA, float* __restrict__ scB)
{
    __shared__ float red[256];
    const int row = blockIdx.x, tid = threadIdx.x;
    const float* src = (blockIdx.y ? mB : mA) + (size_t)row * SEQ;
    float* dst       = (blockIdx.y ? scB : scA) + (size_t)row * SEQ;

    float x0 = src[tid], x1 = src[tid + 256];
    red[tid] = fmaxf(x0, x1);
    __syncthreads();
    for (int s = 128; s > 0; s >>= 1) {
        if (tid < s) red[tid] = fmaxf(red[tid], red[tid + s]);
        __syncthreads();
    }
    const float mx = red[0];
    __syncthreads();
    float e0 = expf(x0 - mx), e1 = expf(x1 - mx);
    red[tid] = e0 + e1;
    __syncthreads();
    for (int s = 128; s > 0; s >>= 1) {
        if (tid < s) red[tid] += red[tid + s];
        __syncthreads();
    }
    const float inv = 1.0f / red[0];
    dst[tid]       = e0 * inv;
    dst[tid + 256] = e1 * inv;
}

// ---------------------------------------------------------------------------
// 6. attention pooling: out[pair][d] = sum_s src[d][s]*score[pair][s]
// ---------------------------------------------------------------------------
__global__ void k_pool(const float* __restrict__ A, const float* __restrict__ B,
                       const float* __restrict__ scA, const float* __restrict__ scB,
                       float* __restrict__ out)
{
    __shared__ float s[SEQ];
    const int pair = blockIdx.x, tid = threadIdx.x;
    const int a = pair >> 4, b = pair & 15;
    const float* src;
    const float* sc;
    float* o;
    if (blockIdx.y == 0) {
        src = A + (size_t)a * DIM * SEQ; sc = scA + (size_t)pair * SEQ;
        o = out + (size_t)pair * DIM;
    } else {
        src = B + (size_t)b * DIM * SEQ; sc = scB + (size_t)pair * SEQ;
        o = out + (size_t)NA * NB * DIM + (size_t)pair * DIM;
    }
    s[tid] = sc[tid];
    s[tid + 256] = sc[tid + 256];
    __syncthreads();

    const int wave = tid >> 5, lane = tid & 31;
    for (int d0 = 0; d0 < DIM; d0 += 8) {
        const int d = d0 + wave;
        float sum = 0.f;
        for (int t = lane; t < SEQ; t += 32)
            sum += src[(size_t)d * SEQ + t] * s[t];
#pragma unroll
        for (int off = 16; off > 0; off >>= 1)
            sum += __shfl_xor(sum, off, 32);
        if (lane == 0) o[d] = sum;
    }
}

// ---------------------------------------------------------------------------
// launch
// ---------------------------------------------------------------------------
extern "C" void kernel_launch(void* const* d_in, const int* in_sizes, int n_in,
                              void* d_out, int out_size, void* d_ws, size_t ws_size,
                              hipStream_t stream)
{
    const float* inA = (const float*)d_in[0];   // (16, 512, 512)
    const float* inB = (const float*)d_in[1];   // (16, 512, 512)
    const float* msk = (const float*)d_in[2];   // (16, 512, 512)
    const float* U   = (const float*)d_in[3];   // (512, 512)
    float* out = (float*)d_out;

    unsigned short* At  = (unsigned short*)d_ws;                  // 16*512*512 bf16
    unsigned int*   Bsw = (unsigned int*)(At + (size_t)NA * SEQ * DIM);   // 16*512*256 u32
    unsigned int*   Usw = Bsw + (size_t)NB * SEQ * (DIM / 2);     // 512*256 u32
    unsigned short* Pr  = (unsigned short*)(Usw + (size_t)DIM * (DIM / 2)); // 16*512*512 bf16
    float* mA  = (float*)(Pr + (size_t)NA * SEQ * DIM);           // 16*16*512 f32
    float* mB  = mA + NA * NB * SEQ;
    float* scA = mB + NA * NB * SEQ;
    float* scB = scA + NA * NB * SEQ;

    // 1. precision conversion (+ transpose of A, n-major k-pair swizzle of B, U)
    k_transposeA<<<dim3(16, 16, 16), dim3(32, 8), 0, stream>>>(inA, At);
    {
        int nW = NB * SEQ * (DIM / 2);
        k_swzconvert<<<(nW + 255) / 256, 256, 0, stream>>>(inB, Bsw, nW);
    }
    {
        int nW = DIM * (DIM / 2);
        k_swzconvert<<<(nW + 255) / 256, 256, 0, stream>>>(U, Usw, nW);
    }

    // 2. -inf init of max-pool logit buffers (mA, mB contiguous)
    k_fillneg<<<(2 * NA * NB * SEQ) / 256, 256, 0, stream>>>(mA, 2 * NA * NB * SEQ);

    // 3. proj = A^T @ U  (bf16 WMMA, TDM load + TDM store)
    k_gemm_proj<<<dim3(16, NA), 256, 0, stream>>>(At, Usw, Pr);

    // 4. align GEMM + fused tanh/mask/max-pool (bf16 WMMA, TDM-fed)
    k_align<<<dim3(16, NB, NA), 256, 0, stream>>>(Pr, Bsw, msk, mA, mB);

    // 5. softmax
    k_softmax<<<dim3(NA * NB, 2), 256, 0, stream>>>(mA, mB, scA, scB);

    // 6. attention pooling
    k_pool<<<dim3(NA * NB, 2), 256, 0, stream>>>(inA, inB, scA, scB, out);
}
